// PointerAttention_12111807775132
// MI455X (gfx1250) — matrix operand order repaired and checked by reading, not compiled, not run on hardware
//
#include <hip/hip_runtime.h>

// Problem sizes (match reference)
constexpr int B_ = 128, S_ = 2048, D_ = 512, H_ = 512;

typedef __attribute__((ext_vector_type(4)))  float        f4;
typedef __attribute__((ext_vector_type(4)))  unsigned int u4;
typedef __attribute__((ext_vector_type(8)))  float        v8f;
typedef __attribute__((ext_vector_type(16))) __bf16       v16bf;

union V16BF { v16bf v; unsigned int u[8]; u4 q[2]; };

__device__ __forceinline__ unsigned int f2bf_bits(float f) {
    unsigned int u = __builtin_bit_cast(unsigned int, f);
    return (u + 0x7FFFu + ((u >> 16) & 1u)) >> 16;   // RNE truncate to bf16
}
__device__ __forceinline__ float bf2f(unsigned int hi) {
    return __builtin_bit_cast(float, hi << 16);
}
__device__ __forceinline__ void cvt_hl(float x, float y, unsigned int& hw, unsigned int& lw) {
    unsigned int hx = f2bf_bits(x), hy = f2bf_bits(y);
    float lx = x - bf2f(hx), ly = y - bf2f(hy);
    hw = hx | (hy << 16);
    lw = f2bf_bits(lx) | (f2bf_bits(ly) << 16);
}
__device__ __forceinline__ float fast_tanh(float x) {
    // tanh(x) = 1 - 2/(exp(2x)+1); exact saturation at +-inf of exp2
    float e = __builtin_amdgcn_exp2f(x * 2.88539008177792681f); // 2*log2(e)
    return 1.0f - 2.0f * __builtin_amdgcn_rcpf(e + 1.0f);
}

// ---------------------------------------------------------------------------
// K0: pre-swizzle W_ctx into the WMMA B-matrix lane layout, split bf16 hi/lo.
// Flat element i = ((ht*16 + kit)*32 + lane)*16 + e maps to
//   h = ht*16 + (lane&15),  k = kit*32 + (lane>>4)*8 + (e&7) + (e>>3)*16
// ---------------------------------------------------------------------------
__global__ __launch_bounds__(256) void k0_swizzle(const float* __restrict__ W_ctx,
                                                  unsigned short* __restrict__ Wbh,
                                                  unsigned short* __restrict__ Wbl) {
    int i = blockIdx.x * 256 + threadIdx.x;            // 0 .. 262143
    int e = i & 15, lane = (i >> 4) & 31, kit = (i >> 9) & 15, ht = i >> 13;
    int h = ht * 16 + (lane & 15);
    int k = kit * 32 + (lane >> 4) * 8 + (e & 7) + (e >> 3) * 16;
    float v = W_ctx[(size_t)h * D_ + k];
    unsigned int hb = f2bf_bits(v);
    float lo = v - bf2f(hb);
    Wbh[i] = (unsigned short)hb;
    Wbl[i] = (unsigned short)f2bf_bits(lo);
}

// ---------------------------------------------------------------------------
// K1: inp = input @ W_in^T + b_in   [B,H]  (tiny, fp32 exact)
// ---------------------------------------------------------------------------
__global__ __launch_bounds__(256) void k1_inp(const float* __restrict__ input,
                                              const float* __restrict__ W_in,
                                              const float* __restrict__ b_in,
                                              float* __restrict__ inp) {
    __shared__ float xr[D_];
    int b = blockIdx.x, tid = threadIdx.x;
    for (int d = tid; d < D_; d += 256) xr[d] = input[(size_t)b * D_ + d];
    __syncthreads();
    for (int h = tid; h < H_; h += 256) {
        const f4* wr = (const f4*)(W_in + (size_t)h * D_);
        float acc = b_in[h];
        for (int q = 0; q < D_ / 4; ++q) {
            f4 w = wr[q];
            acc += w.x * xr[4*q] + w.y * xr[4*q+1] + w.z * xr[4*q+2] + w.w * xr[4*q+3];
        }
        inp[(size_t)b * H_ + h] = acc;
    }
}

// ---------------------------------------------------------------------------
// K2: main fused GEMM + tanh-attention-logit kernel.
// Grid (S/64, B); 256 threads = 8 waves. Each WG: 64 s-rows x all 512 h.
// Wave w: 4 s-subtiles x 4 h-tiles of 16x16, K=512 in steps of 32.
// bf16x3 split WMMA: acc += Ah*Bh + Al*Bh + Ah*Bl  (near-fp32 precision).
// Epilogue: att[b,s] = sum_h V[h]*tanh(inp[b,h]+b_ctx[h]+ctx[s,h]),
// reduced deterministically (shuffle butterfly + fixed-order LDS sum).
// ---------------------------------------------------------------------------
__global__ __launch_bounds__(256) void k2_ctx_att(const float* __restrict__ context,
                                                  const unsigned short* __restrict__ Wbh,
                                                  const unsigned short* __restrict__ Wbl,
                                                  const float* __restrict__ inp,
                                                  const float* __restrict__ b_ctx,
                                                  const float* __restrict__ V,
                                                  float* __restrict__ att) {
    __shared__ float lds_add[H_];
    __shared__ float lds_scale[H_];
    __shared__ float part[8][64];

    const int tid = threadIdx.x, lane = tid & 31, w = tid >> 5;
    const int b = blockIdx.y, s0 = blockIdx.x * 64;
    const int n = lane & 15, hi = lane >> 4;

    for (int h = tid; h < H_; h += 256) {
        lds_add[h]   = inp[(size_t)b * H_ + h] + b_ctx[h];
        lds_scale[h] = V[h];
    }
    __syncthreads();

    const float* ctxb = context + ((size_t)b * S_ + s0) * D_;

    v8f zero = {0.f, 0.f, 0.f, 0.f, 0.f, 0.f, 0.f, 0.f};
    v8f acc[4][4];
    #pragma unroll
    for (int ss = 0; ss < 4; ++ss)
        #pragma unroll
        for (int j = 0; j < 4; ++j) acc[ss][j] = zero;

    for (int kit = 0; kit < 16; ++kit) {
        const int k1 = kit * 32 + hi * 8;
        V16BF Ah[4], Al[4];
        #pragma unroll
        for (int ss = 0; ss < 4; ++ss) {
            const float* row = ctxb + (size_t)(ss * 16 + n) * D_ + k1;
            f4 a0 = *(const f4*)(row);
            f4 a1 = *(const f4*)(row + 4);
            f4 a2 = *(const f4*)(row + 16);
            f4 a3 = *(const f4*)(row + 20);
            if (kit < 15) __builtin_prefetch(row + 32, 0, 1);
            cvt_hl(a0.x, a0.y, Ah[ss].u[0], Al[ss].u[0]);
            cvt_hl(a0.z, a0.w, Ah[ss].u[1], Al[ss].u[1]);
            cvt_hl(a1.x, a1.y, Ah[ss].u[2], Al[ss].u[2]);
            cvt_hl(a1.z, a1.w, Ah[ss].u[3], Al[ss].u[3]);
            cvt_hl(a2.x, a2.y, Ah[ss].u[4], Al[ss].u[4]);
            cvt_hl(a2.z, a2.w, Ah[ss].u[5], Al[ss].u[5]);
            cvt_hl(a3.x, a3.y, Ah[ss].u[6], Al[ss].u[6]);
            cvt_hl(a3.z, a3.w, Ah[ss].u[7], Al[ss].u[7]);
        }
        #pragma unroll
        for (int j = 0; j < 4; ++j) {
            const int ht = w * 4 + j;
            const size_t bidx = (((size_t)ht * 16 + kit) * 32 + lane) * 16;
            V16BF Bh, Bl;
            const u4* bph = (const u4*)(Wbh + bidx);
            const u4* bpl = (const u4*)(Wbl + bidx);
            Bh.q[0] = bph[0]; Bh.q[1] = bph[1];
            Bl.q[0] = bpl[0]; Bl.q[1] = bpl[1];
            #pragma unroll
            for (int ss = 0; ss < 4; ++ss) {
                acc[ss][j] = __builtin_amdgcn_wmma_f32_16x16x32_bf16(
                    false, Ah[ss].v, false, Bh.v, (short)0, acc[ss][j], false, false);
                acc[ss][j] = __builtin_amdgcn_wmma_f32_16x16x32_bf16(
                    false, Al[ss].v, false, Bh.v, (short)0, acc[ss][j], false, false);
                acc[ss][j] = __builtin_amdgcn_wmma_f32_16x16x32_bf16(
                    false, Ah[ss].v, false, Bl.v, (short)0, acc[ss][j], false, false);
            }
        }
    }

    // Epilogue: per wave, per-row partial sums over its 64 h-columns.
    #pragma unroll
    for (int ss = 0; ss < 4; ++ss) {
        float ts[8];
        #pragma unroll
        for (int v = 0; v < 8; ++v) ts[v] = 0.f;
        #pragma unroll
        for (int j = 0; j < 4; ++j) {
            const int h = w * 64 + j * 16 + n;
            const float addv = lds_add[h], sc = lds_scale[h];
            #pragma unroll
            for (int v = 0; v < 8; ++v)
                ts[v] += sc * fast_tanh(acc[ss][j][v] + addv);
        }
        #pragma unroll
        for (int v = 0; v < 8; ++v) {
            float t = ts[v];
            #pragma unroll
            for (int m = 1; m < 16; m <<= 1) t += __shfl_xor(t, m, 32);
            if (n == 0) part[w][ss * 16 + hi * 8 + v] = t;  // C/D layout: row = v + hi*8
        }
    }
    __syncthreads();
    if (tid < 64) {
        float s = 0.f;
        #pragma unroll
        for (int ww = 0; ww < 8; ++ww) s += part[ww][tid];  // fixed order -> deterministic
        att[(size_t)b * S_ + s0 + tid] = s;
    }
}

// ---------------------------------------------------------------------------
// K3: masked softmax over S per row b.  mask==true -> -inf (excluded).
// ---------------------------------------------------------------------------
__global__ __launch_bounds__(256) void k3_softmax(const float* __restrict__ att,
                                                  const unsigned char* __restrict__ mask,
                                                  float* __restrict__ alpha) {
    __shared__ float red[256];
    const int b = blockIdx.x, tid = threadIdx.x;
    float a[8];
    float mx = -__builtin_inff();
    #pragma unroll
    for (int i = 0; i < 8; ++i) {
        const size_t idx = (size_t)b * S_ + tid + i * 256;
        float v = att[idx];
        v = mask[idx] ? -__builtin_inff() : v;
        a[i] = v;
        mx = fmaxf(mx, v);
    }
    red[tid] = mx; __syncthreads();
    for (int off = 128; off > 0; off >>= 1) {
        if (tid < off) red[tid] = fmaxf(red[tid], red[tid + off]);
        __syncthreads();
    }
    mx = red[0]; __syncthreads();

    float sum = 0.f;
    #pragma unroll
    for (int i = 0; i < 8; ++i) {
        a[i] = __builtin_amdgcn_exp2f((a[i] - mx) * 1.44269504088896340f); // exp(-inf)=0
        sum += a[i];
    }
    red[tid] = sum; __syncthreads();
    for (int off = 128; off > 0; off >>= 1) {
        if (tid < off) red[tid] += red[tid + off];
        __syncthreads();
    }
    const float inv = 1.0f / red[0];
    #pragma unroll
    for (int i = 0; i < 8; ++i)
        alpha[(size_t)b * S_ + tid + i * 256] = a[i] * inv;
}

// ---------------------------------------------------------------------------
// K4: partial c_bar[b][chunk][d] = sum_{s in chunk} alpha[b,s]*context[b,s,d]
// Skips rows with alpha==0 (masked) -> ~halves second context read.
// ---------------------------------------------------------------------------
__global__ __launch_bounds__(256) void k4_partial(const float* __restrict__ context,
                                                  const float* __restrict__ alpha,
                                                  float* __restrict__ partial) {
    __shared__ float la[128];
    const int b = blockIdx.y, c = blockIdx.x, tid = threadIdx.x;
    if (tid < 128) la[tid] = alpha[(size_t)b * S_ + c * 128 + tid];
    __syncthreads();
    float acc0 = 0.f, acc1 = 0.f;
    const float* base = context + ((size_t)b * S_ + (size_t)c * 128) * D_;
    for (int s = 0; s < 128; ++s) {
        const float av = la[s];
        if (av == 0.0f) continue;            // uniform branch across block
        const float* row = base + (size_t)s * D_;
        acc0 += av * row[tid];
        acc1 += av * row[tid + 256];
    }
    float* p = partial + ((size_t)b * 16 + c) * D_;
    p[tid] = acc0;
    p[tid + 256] = acc1;
}

// ---------------------------------------------------------------------------
// K5: hidden[b,h] = sum_d W_ctx[h,d] * c_bar[b,d] + b_ctx[h]
// (c_bar = fixed-order reduction of the 16 chunk partials)
// ---------------------------------------------------------------------------
__global__ __launch_bounds__(256) void k5_hidden(const float* __restrict__ partial,
                                                 const float* __restrict__ W_ctx,
                                                 const float* __restrict__ b_ctx,
                                                 float* __restrict__ hidden) {
    __shared__ float cb[D_];
    const int b = blockIdx.x, tid = threadIdx.x;
    for (int d = tid; d < D_; d += 256) {
        float s = 0.f;
        #pragma unroll
        for (int c = 0; c < 16; ++c) s += partial[((size_t)b * 16 + c) * D_ + d];
        cb[d] = s;
    }
    __syncthreads();
    for (int h = tid; h < H_; h += 256) {
        const f4* wr = (const f4*)(W_ctx + (size_t)h * D_);
        float acc = b_ctx[h];
        for (int q = 0; q < D_ / 4; ++q) {
            f4 w = wr[q];
            acc += w.x * cb[4*q] + w.y * cb[4*q+1] + w.z * cb[4*q+2] + w.w * cb[4*q+3];
        }
        hidden[(size_t)b * H_ + h] = acc;
    }
}

// ---------------------------------------------------------------------------
extern "C" void kernel_launch(void* const* d_in, const int* in_sizes, int n_in,
                              void* d_out, int out_size, void* d_ws, size_t ws_size,
                              hipStream_t stream) {
    const float*         input   = (const float*)d_in[0];
    const float*         context = (const float*)d_in[1];
    const unsigned char* mask    = (const unsigned char*)d_in[2]; // jax bool = 1 byte
    const float*         W_in    = (const float*)d_in[3];
    const float*         b_in    = (const float*)d_in[4];
    const float*         W_ctx   = (const float*)d_in[5];
    const float*         b_ctx   = (const float*)d_in[6];
    const float*         V       = (const float*)d_in[7];

    float* out    = (float*)d_out;
    float* hidden = out;                    // [B,H] = 65536
    float* alpha  = out + B_ * H_;          // [B,S] = 262144

    // workspace layout (float offsets)
    float*          ws      = (float*)d_ws;
    float*          inp     = ws;                                   //  65536 f
    float*          att     = ws + 65536;                           // 262144 f
    unsigned short* Wbh     = (unsigned short*)(ws + 327680);       // 262144 u16
    unsigned short* Wbl     = (unsigned short*)(ws + 458752);       // 262144 u16
    float*          partial = ws + 589824;                          // 128*16*512 f
    (void)in_sizes; (void)n_in; (void)out_size; (void)ws_size;

    k0_swizzle<<<dim3((H_ * D_) / 256), dim3(256), 0, stream>>>(W_ctx, Wbh, Wbl);
    k1_inp<<<dim3(B_), dim3(256), 0, stream>>>(input, W_in, b_in, inp);
    k2_ctx_att<<<dim3(S_ / 64, B_), dim3(256), 0, stream>>>(context, Wbh, Wbl, inp, b_ctx, V, att);
    k3_softmax<<<dim3(B_), dim3(256), 0, stream>>>(att, mask, alpha);
    k4_partial<<<dim3(16, B_), dim3(256), 0, stream>>>(context, alpha, partial);
    k5_hidden<<<dim3(B_), dim3(256), 0, stream>>>(partial, W_ctx, b_ctx, hidden);
}
